// ISTEncoder_64647847739809
// MI455X (gfx1250) — matrix operand order repaired and checked by reading, not compiled, not run on hardware
//
#include <hip/hip_runtime.h>
#include <cmath>

#define NTX 50000
#define NBD 5000
#define ETT 500000
#define ETB 50000
#define EBT 50000

typedef __attribute__((ext_vector_type(16))) _Float16 v16h;
typedef __attribute__((ext_vector_type(8)))  float    v8f;
typedef __attribute__((ext_vector_type(4)))  float    v4f;

// ---------------- small device helpers ----------------
__device__ __forceinline__ float gelu_f(float x) {
  return 0.5f * x * (1.0f + erff(x * 0.70710678118654752f));
}
__device__ __forceinline__ void atomicMaxF(float* a, float v) {
  if (v >= 0.0f) atomicMax((int*)a, __float_as_int(v));
  else           atomicMin((unsigned int*)a, __float_as_uint(v));
}
__device__ __forceinline__ void atomicMinF(float* a, float v) {
  if (v >= 0.0f) atomicMin((int*)a, __float_as_int(v));
  else           atomicMax((unsigned int*)a, __float_as_uint(v));
}

// ---------------- generic fill ----------------
__global__ void fill_kernel(float* __restrict__ p, long long n, float v) {
  long long i = (long long)blockIdx.x * blockDim.x + threadIdx.x;
  if (i < n) p[i] = v;
}

// ---------------- per-graph pos min/max ----------------
__global__ void posminmax_kernel(const float* __restrict__ pos, const int* __restrict__ batch,
                                 int N, float* __restrict__ mn, float* __restrict__ mx) {
  int i = blockIdx.x * blockDim.x + threadIdx.x;
  if (i >= N) return;
  int g = batch[i];
  float px = pos[2 * i], py = pos[2 * i + 1];
  atomicMinF(&mn[g * 2 + 0], px); atomicMaxF(&mx[g * 2 + 0], px);
  atomicMinF(&mn[g * 2 + 1], py); atomicMaxF(&mx[g * 2 + 1], py);
}

// ---------------- positional MLP (per coordinate) ----------------
__device__ void pos_embed_dev(float pn, const float* __restrict__ sW1,
                              const float* __restrict__ b1, const float* __restrict__ W2,
                              const float* __restrict__ b2, float* out8) {
  float h1[8];
#pragma unroll
  for (int k = 0; k < 8; k++) h1[k] = b1[k];
  for (int j = 0; j < 128; j++) {
    float f = __expf(-9.2103403719761836f * (float)j * (1.0f / 128.0f));
    float ang = pn * f;
    float cj = __cosf(ang), sj = __sinf(ang);
#pragma unroll
    for (int k = 0; k < 8; k++)
      h1[k] = fmaf(cj, sW1[j * 8 + k], fmaf(sj, sW1[(128 + j) * 8 + k], h1[k]));
  }
#pragma unroll
  for (int k = 0; k < 8; k++) { float x = h1[k]; h1[k] = x / (1.0f + __expf(-x)); }
#pragma unroll
  for (int d2 = 0; d2 < 8; d2++) {
    float a = b2[d2];
#pragma unroll
    for (int k = 0; k < 8; k++) a = fmaf(h1[k], W2[k * 8 + d2], a);
    out8[d2] = a;
  }
}

// ---------------- node embedding kernels (write h stride 32, gelu applied) ----------------
__global__ void embed_tx_kernel(const int* __restrict__ xtx, const float* __restrict__ emb,
                                const float* __restrict__ pos, const int* __restrict__ batch,
                                const float* __restrict__ mn, const float* __restrict__ mx,
                                const float* __restrict__ W1, const float* __restrict__ b1,
                                const float* __restrict__ W2, const float* __restrict__ b2,
                                float* __restrict__ h, int N) {
  __shared__ float sW1[2048];
  for (int i = threadIdx.x; i < 2048; i += blockDim.x) sW1[i] = W1[i];
  __syncthreads();
  int i = blockIdx.x * blockDim.x + threadIdx.x;
  if (i >= N) return;
  float out[32];
  const float* e = emb + (size_t)xtx[i] * 16;
#pragma unroll
  for (int c = 0; c < 16; c++) out[c] = e[c];
  int g = batch[i];
#pragma unroll
  for (int d = 0; d < 2; d++) {
    float lo = mn[g * 2 + d], hi = mx[g * 2 + d];
    float pn = (pos[2 * i + d] - lo) / (hi - lo + 1e-8f);
    pos_embed_dev(pn, sW1, b1, W2, b2, &out[16 + 8 * d]);
  }
#pragma unroll
  for (int c = 0; c < 32; c++) h[(size_t)i * 32 + c] = gelu_f(out[c]);
}

__global__ void embed_bd_kernel(const float* __restrict__ xbd, const float* __restrict__ bW,
                                const float* __restrict__ bb, const float* __restrict__ pos,
                                const int* __restrict__ batch, const float* __restrict__ mn,
                                const float* __restrict__ mx, const float* __restrict__ W1,
                                const float* __restrict__ b1, const float* __restrict__ W2,
                                const float* __restrict__ b2, float* __restrict__ h, int N) {
  __shared__ float sW1[2048];
  for (int i = threadIdx.x; i < 2048; i += blockDim.x) sW1[i] = W1[i];
  __syncthreads();
  int i = blockIdx.x * blockDim.x + threadIdx.x;
  if (i >= N) return;
  float out[32];
  const float* xr = xbd + (size_t)i * 32;
#pragma unroll
  for (int c = 0; c < 16; c++) {
    float a = bb[c];
    for (int k = 0; k < 32; k++) a = fmaf(xr[k], bW[k * 16 + c], a);
    out[c] = a;
  }
  int g = batch[i];
#pragma unroll
  for (int d = 0; d < 2; d++) {
    float lo = mn[g * 2 + d], hi = mx[g * 2 + d];
    float pn = (pos[2 * i + d] - lo) / (hi - lo + 1e-8f);
    pos_embed_dev(pn, sW1, b1, W2, b2, &out[16 + 8 * d]);
  }
#pragma unroll
  for (int c = 0; c < 32; c++) h[(size_t)i * 32 + c] = gelu_f(out[c]);
}

// ---------------- WMMA GEMM: Y[N x DOUT] = X[N x DIN] @ W[DIN x DOUT] + bias ----------------
// One wave per 16-row M tile. W staged to LDS as f16 in *fragment order* so each lane's
// 16-half B fragment is one contiguous 32B LDS load (2x ds_load_b128 instead of 16x ds_load_u16).
// A fragments are two contiguous 8-float runs per lane -> 4x global_load_b128 + cvt.
template <int DOUT>
__global__ void gemm_wmma_kernel(const float* __restrict__ X, const float* __restrict__ W,
                                 const float* __restrict__ bias, float* __restrict__ Y,
                                 int N, int DIN) {
  constexpr int NT = DOUT / 16;
  extern __shared__ _Float16 sW[];   // DIN * DOUT halves, fragment-ordered

  // stage + permute: i = ((ks*NT + t)*32 + lane)*16 + e
  int total = DIN * DOUT;
  for (int i = threadIdx.x; i < total; i += blockDim.x) {
    int e  = i & 15;
    int ln = (i >> 4) & 31;
    int t  = (i >> 9) % NT;
    int ks = i / (512 * NT);
    int K  = ks * 32 + 8 * (ln >> 4) + (e & 7) + ((e >> 3) << 4);
    int n  = t * 16 + (ln & 15);
    sW[i] = (_Float16)W[(size_t)K * DOUT + n];
  }
  __syncthreads();

  int lane = threadIdx.x & 31;
  int wave = threadIdx.x >> 5;
  int lh = lane >> 4;          // 0 or 1 (half-wave)
  int ll = lane & 15;
  int mbase = (blockIdx.x * (blockDim.x >> 5) + wave) * 16;

  v8f zero = {0.f, 0.f, 0.f, 0.f, 0.f, 0.f, 0.f, 0.f};
  v8f acc[NT];
#pragma unroll
  for (int t = 0; t < NT; t++) acc[t] = zero;

  int m = mbase + ll;
  if (m >= N) m = 0;           // clamp; padded rows never stored
  const float* xrow = X + (size_t)m * DIN;

  int nks = DIN >> 5;
  for (int ks = 0; ks < nks; ks++) {
    int kb = ks * 32;
    if (kb + 32 < DIN) __builtin_prefetch(xrow + kb + 32, 0, 1);
    // A fragment: runs [kb+8*lh, +8) and [kb+16+8*lh, +8)  (16B-aligned)
    const v4f* p0 = (const v4f*)(xrow + kb + 8 * lh);
    const v4f* p1 = (const v4f*)(xrow + kb + 16 + 8 * lh);
    v4f q0 = p0[0], q1 = p0[1], q2 = p1[0], q3 = p1[1];
    v16h afrag;
#pragma unroll
    for (int e = 0; e < 4; e++) {
      afrag[e]      = (_Float16)q0[e];
      afrag[4 + e]  = (_Float16)q1[e];
      afrag[8 + e]  = (_Float16)q2[e];
      afrag[12 + e] = (_Float16)q3[e];
    }
#pragma unroll
    for (int t = 0; t < NT; t++) {
      const v16h* bp = (const v16h*)(sW + (size_t)(((ks * NT + t) * 32 + lane) * 16));
      v16h bfrag = *bp;
      acc[t] = __builtin_amdgcn_wmma_f32_16x16x32_f16(
          false, afrag, false, bfrag, (short)0, acc[t], false, false);
    }
  }
#pragma unroll
  for (int t = 0; t < NT; t++) {
    int n = t * 16 + ll;
    float bv = bias[n];
#pragma unroll
    for (int r = 0; r < 8; r++) {
      int mm = mbase + r + 8 * lh;         // wave32 f32 D layout
      if (mm < N) Y[(size_t)mm * DOUT + n] = acc[t][r] + bv;
    }
  }
}

// ---------------- GATv2 edge kernels ----------------
__global__ void edge_logits_kernel(const int* __restrict__ src, const int* __restrict__ dst,
                                   int E, const float* __restrict__ xl, const float* __restrict__ xr,
                                   const float* __restrict__ att, float* __restrict__ logits,
                                   float* __restrict__ mbuf) {
  int t = blockIdx.x * blockDim.x + threadIdx.x;
  if (t >= E * 3) return;
  int e = t / 3, h = t % 3;
  int s = src[e], d = dst[e];
  const float* pl = xl + (size_t)s * 96 + h * 32;
  const float* pr = xr + (size_t)d * 96 + h * 32;
  const float* pa = att + h * 32;
  float acc = 0.f;
#pragma unroll
  for (int c = 0; c < 32; c++) {
    float v = pl[c] + pr[c];
    v = (v > 0.f) ? v : 0.2f * v;          // leaky_relu(0.2)
    acc = fmaf(v, pa[c], acc);
  }
  logits[t] = acc;
  atomicMaxF(&mbuf[d * 3 + h], acc);
}

__global__ void edge_softmax_kernel(const int* __restrict__ dst, int E,
                                    float* __restrict__ logits, const float* __restrict__ mbuf,
                                    float* __restrict__ sbuf) {
  int t = blockIdx.x * blockDim.x + threadIdx.x;
  if (t >= E * 3) return;
  int e = t / 3, h = t % 3;
  int d = dst[e];
  float a = __expf(logits[t] - mbuf[d * 3 + h]);
  logits[t] = a;                            // store unnormalized weight
  atomicAdd(&sbuf[d * 3 + h], a);
}

__global__ void edge_scatter_kernel(const int* __restrict__ src, const int* __restrict__ dst,
                                    int E, const float* __restrict__ xl,
                                    const float* __restrict__ a, float* __restrict__ acc) {
  int t = blockIdx.x * blockDim.x + threadIdx.x;
  long long tot = (long long)E * 96;
  if (t >= tot) return;
  int e = t / 96, c = t % 96, h = c >> 5;
  int s = src[e], d = dst[e];
  float v = a[e * 3 + h] * xl[(size_t)s * 96 + c];
  atomicAdd(&acc[(size_t)d * 96 + c], v);
}

// out[dst] = acc/(s+eps) + bias, write or accumulate into message buffer
__global__ void finalize_kernel(const float* __restrict__ acc, const float* __restrict__ sbuf,
                                const float* __restrict__ bias, float* __restrict__ hn,
                                int N, int add) {
  long long t = (long long)blockIdx.x * blockDim.x + threadIdx.x;
  if (t >= (long long)N * 96) return;
  int n = (int)(t / 96), c = (int)(t % 96), h = c >> 5;
  float v = acc[t] / (sbuf[n * 3 + h] + 1e-16f) + bias[c];
  if (add) hn[t] += v; else hn[t] = v;
}

__global__ void gelu_kernel(float* __restrict__ p, long long n) {
  long long i = (long long)blockIdx.x * blockDim.x + threadIdx.x;
  if (i < n) p[i] = gelu_f(p[i]);
}

__global__ void l2norm_kernel(const float* __restrict__ in, float* __restrict__ out, int N) {
  int i = blockIdx.x * blockDim.x + threadIdx.x;
  if (i >= N) return;
  float v[32], ss = 0.f;
#pragma unroll
  for (int c = 0; c < 32; c++) { v[c] = in[(size_t)i * 32 + c]; ss = fmaf(v[c], v[c], ss); }
  float nrm = fmaxf(sqrtf(ss), 1e-12f);
#pragma unroll
  for (int c = 0; c < 32; c++) out[(size_t)i * 32 + c] = v[c] / nrm;
}

// ---------------- host orchestration ----------------
extern "C" void kernel_launch(void* const* d_in, const int* in_sizes, int n_in,
                              void* d_out, int out_size, void* d_ws, size_t ws_size,
                              hipStream_t stream) {
  (void)in_sizes; (void)n_in; (void)out_size; (void)ws_size;

  const int*   x_tx    = (const int*)d_in[0];
  const float* x_bd    = (const float*)d_in[1];
  const float* pos_tx  = (const float*)d_in[2];
  const float* pos_bd  = (const float*)d_in[3];
  const int*   batch_tx= (const int*)d_in[4];
  const int*   batch_bd= (const int*)d_in[5];
  const int*   edge_tt = (const int*)d_in[6];          // [2, ETT] row-major
  const int*   tt_src  = edge_tt;
  const int*   tt_dst  = edge_tt + ETT;
  const int*   tb_src  = (const int*)d_in[7];
  const int*   tb_dst  = (const int*)d_in[8];
  const int*   bt_src  = (const int*)d_in[9];
  const int*   bt_dst  = (const int*)d_in[10];
  auto F = [&](int i) { return (const float*)d_in[i]; };
  const float* emb_tx = F(11);
  const float* bdW = F(12); const float* bdB = F(13);
  const float* pW1 = F(14); const float* pB1 = F(15);
  const float* pW2 = F(16); const float* pB2 = F(17);
  const float* lastTxW = F(108); const float* lastTxB = F(109);
  const float* lastBdW = F(110); const float* lastBdB = F(111);

  // workspace carve-up (floats)
  float* ws = (float*)d_ws;
  size_t off = 0;
  auto take = [&](size_t n) { float* p = ws + off; off += n; return p; };
  float* hA_tx  = take((size_t)NTX * 96);
  float* hA_bd  = take((size_t)NBD * 96);
  float* hB_tx  = take((size_t)NTX * 96);
  float* hB_bd  = take((size_t)NBD * 96);
  float* xl     = take((size_t)NTX * 96);
  float* xr     = take((size_t)NTX * 96);
  float* logits = take((size_t)ETT * 3);
  float* mbuf   = take((size_t)NTX * 3);
  float* sbuf   = take((size_t)NTX * 3);
  float* accb   = take((size_t)NTX * 96);
  float* mn_tx  = take(8); float* mx_tx = take(8);
  float* mn_bd  = take(8); float* mx_bd = take(8);

  const float INF = __builtin_huge_valf();

  // per-graph pos min/max
  fill_kernel<<<1, 64, 0, stream>>>(mn_tx, 8, INF);
  fill_kernel<<<1, 64, 0, stream>>>(mx_tx, 8, -INF);
  fill_kernel<<<1, 64, 0, stream>>>(mn_bd, 8, INF);
  fill_kernel<<<1, 64, 0, stream>>>(mx_bd, 8, -INF);
  posminmax_kernel<<<(NTX + 255) / 256, 256, 0, stream>>>(pos_tx, batch_tx, NTX, mn_tx, mx_tx);
  posminmax_kernel<<<(NBD + 255) / 256, 256, 0, stream>>>(pos_bd, batch_bd, NBD, mn_bd, mx_bd);

  // initial embeddings (stride 32, GELU applied)
  embed_tx_kernel<<<(NTX + 127) / 128, 128, 0, stream>>>(x_tx, emb_tx, pos_tx, batch_tx,
      mn_tx, mx_tx, pW1, pB1, pW2, pB2, hA_tx, NTX);
  embed_bd_kernel<<<(NBD + 127) / 128, 128, 0, stream>>>(x_bd, bdW, bdB, pos_bd, batch_bd,
      mn_bd, mx_bd, pW1, pB1, pW2, pB2, hA_bd, NBD);

  auto run_gat = [&](const float* hsrc, int nsrc, const float* hdst, int ndst, int din,
                     const int* esrc, const int* edst, int E,
                     const float* Wl, const float* bl, const float* Wr, const float* br,
                     const float* att, const float* bias, float* hn, int add) {
    size_t shm = (size_t)din * 96 * sizeof(_Float16);
    gemm_wmma_kernel<96><<<(nsrc + 63) / 64, 128, shm, stream>>>(hsrc, Wl, bl, xl, nsrc, din);
    gemm_wmma_kernel<96><<<(ndst + 63) / 64, 128, shm, stream>>>(hdst, Wr, br, xr, ndst, din);
    fill_kernel<<<(ndst * 3 + 255) / 256, 256, 0, stream>>>(mbuf, (long long)ndst * 3, -INF);
    fill_kernel<<<(ndst * 3 + 255) / 256, 256, 0, stream>>>(sbuf, (long long)ndst * 3, 0.f);
    long long na = (long long)ndst * 96;
    fill_kernel<<<(unsigned)((na + 255) / 256), 256, 0, stream>>>(accb, na, 0.f);
    int t1 = E * 3;
    edge_logits_kernel<<<(t1 + 255) / 256, 256, 0, stream>>>(esrc, edst, E, xl, xr, att, logits, mbuf);
    edge_softmax_kernel<<<(t1 + 255) / 256, 256, 0, stream>>>(edst, E, logits, mbuf, sbuf);
    long long t2 = (long long)E * 96;
    edge_scatter_kernel<<<(unsigned)((t2 + 255) / 256), 256, 0, stream>>>(esrc, edst, E, xl, logits, accb);
    finalize_kernel<<<(unsigned)((na + 255) / 256), 256, 0, stream>>>(accb, sbuf, bias, hn, ndst, add);
  };

  float* hc_tx = hA_tx; float* hc_bd = hA_bd;
  float* hn_tx = hB_tx; float* hn_bd = hB_bd;
  int din = 32;
  for (int l = 0; l < 5; l++) {
    int p = 18 + l * 18;   // leaves per layer: tt(6), tb(6), bt(6)
    // tx <- tx (tt)
    run_gat(hc_tx, NTX, hc_tx, NTX, din, tt_src, tt_dst, ETT,
            F(p + 0), F(p + 1), F(p + 2), F(p + 3), F(p + 4), F(p + 5), hn_tx, 0);
    // tx <- bd (bt)
    run_gat(hc_bd, NBD, hc_tx, NTX, din, bt_src, bt_dst, EBT,
            F(p + 12), F(p + 13), F(p + 14), F(p + 15), F(p + 16), F(p + 17), hn_tx, 1);
    // bd <- tx (tb)
    run_gat(hc_tx, NTX, hc_bd, NBD, din, tb_src, tb_dst, ETB,
            F(p + 6), F(p + 7), F(p + 8), F(p + 9), F(p + 10), F(p + 11), hn_bd, 0);
    long long ntx = (long long)NTX * 96, nbd = (long long)NBD * 96;
    gelu_kernel<<<(unsigned)((ntx + 255) / 256), 256, 0, stream>>>(hn_tx, ntx);
    gelu_kernel<<<(unsigned)((nbd + 255) / 256), 256, 0, stream>>>(hn_bd, nbd);
    float* t;
    t = hc_tx; hc_tx = hn_tx; hn_tx = t;
    t = hc_bd; hc_bd = hn_bd; hn_bd = t;
    din = 96;
  }

  // final projection 96->32 (WMMA) then L2 normalize into d_out (tx then bd)
  size_t shm32 = (size_t)96 * 32 * sizeof(_Float16);
  gemm_wmma_kernel<32><<<(NTX + 63) / 64, 128, shm32, stream>>>(hc_tx, lastTxW, lastTxB, xl, NTX, 96);
  gemm_wmma_kernel<32><<<(NBD + 63) / 64, 128, shm32, stream>>>(hc_bd, lastBdW, lastBdB, xr, NBD, 96);
  float* out = (float*)d_out;
  l2norm_kernel<<<(NTX + 255) / 256, 256, 0, stream>>>(xl, out, NTX);
  l2norm_kernel<<<(NBD + 255) / 256, 256, 0, stream>>>(xr, out + (size_t)NTX * 32, NBD);
}
